// MambaDoc_10402410791616
// MI455X (gfx1250) — matrix operand order repaired and checked by reading, not compiled, not run on hardware
//
#include <hip/hip_runtime.h>
#include <math.h>

// ---------------- model dims ----------------
#define BB    4
#define LL    4096
#define NE    64
#define DOUTc 64
#define DI    128
#define NLc   2
#define HH    2
#define NN    128
#define CONVc 4
#define DINc  256     // 2*DI
#define PP    128     // DIN/H
#define CCHc  512     // DIN + 2*N
#define DPROJc 770    // full in_proj width (770 = 768 + 2 dt cols)
#define ZXW   768     // WMMA'd part of in_proj (z + xBC), multiple of 64
#define MM    (BB*LL) // 16384 token rows
#define QC    32      // scan chunk length

typedef __attribute__((ext_vector_type(16))) __bf16 v16bf;
typedef __attribute__((ext_vector_type(8)))  float  v8f;

#define DEV static __device__ __forceinline__

DEV v8f wmma_bf16(v16bf a, v16bf b, v8f c) {
  // D = A(16x32 bf16) * B(32x16 bf16) + C(16x16 f32)
  return __builtin_amdgcn_wmma_f32_16x16x32_bf16(false, a, false, b, (short)0, c,
                                                 false, false);
}

// CDNA5 wave32 16-bit A fragment K map: lanes0-15: K 0-7 (v0-3), 16-23 (v4-7);
// lanes16-31: K 8-15, 24-31.
DEV int ka_map(int e, int half) { return (e < 8) ? (e + 8 * half) : (e + 8 + 8 * half); }
// 16-bit B fragment K map: same as A (lanes0-15: K 0-7, 16-23; lanes16-31: K 8-15, 24-31).
DEV int kb_map(int e, int half) { return (e < 8) ? (e + 8 * half) : (e + 8 + 8 * half); }

// A fragment from f32 row-major [M x K], cvt to bf16
DEV v16bf frag_a_f32(const float* p, int ld, int r0, int k0) {
  int lane = threadIdx.x & 31, half = lane >> 4, m = r0 + (lane & 15);
  v16bf a;
#pragma unroll
  for (int e = 0; e < 16; ++e)
    a[e] = (__bf16)p[(size_t)m * ld + k0 + ka_map(e, half)];
  return a;
}
// A fragment from bf16 row-major
DEV v16bf frag_a_bf16(const __bf16* p, int ld, int r0, int k0) {
  int lane = threadIdx.x & 31, half = lane >> 4, m = r0 + (lane & 15);
  v16bf a;
#pragma unroll
  for (int e = 0; e < 16; ++e) a[e] = p[m * ld + k0 + ka_map(e, half)];
  return a;
}
// A fragment from bf16 row-major, scaled per-row (sc indexed by local row)
DEV v16bf frag_a_bf16_rowscale(const __bf16* p, int ld, int r0, int k0, const float* sc) {
  int lane = threadIdx.x & 31, half = lane >> 4, m = r0 + (lane & 15);
  float s = sc[m];  // lane-invariant across elements: one DS read
  v16bf a;
#pragma unroll
  for (int e = 0; e < 16; ++e)
    a[e] = (__bf16)((float)p[m * ld + k0 + ka_map(e, half)] * s);
  return a;
}
// A(m,k) = src[k*ld + m] * sc[k]  (transposed bf16 source with per-k scale)
DEV v16bf frag_at_bf16_kscale(const __bf16* p, int ld, int r0, int k0, const float* sc) {
  int lane = threadIdx.x & 31, half = lane >> 4, m = r0 + (lane & 15);
  v16bf a;
#pragma unroll
  for (int e = 0; e < 16; ++e) {
    int k = k0 + ka_map(e, half);
    a[e] = (__bf16)((float)p[k * ld + m] * sc[k]);
  }
  return a;
}
// B(k,n) from bf16 row-major [K x N]
DEV v16bf frag_b_bf16(const __bf16* p, int ld, int k0, int n0) {
  int lane = threadIdx.x & 31, half = lane >> 4, n = n0 + (lane & 15);
  v16bf b;
#pragma unroll
  for (int e = 0; e < 16; ++e) b[e] = p[(k0 + kb_map(e, half)) * ld + n];
  return b;
}
// B(k,n) from f32 row-major [K x N], cvt bf16
DEV v16bf frag_b_f32(const float* p, int ld, int k0, int n0) {
  int lane = threadIdx.x & 31, half = lane >> 4, n = n0 + (lane & 15);
  v16bf b;
#pragma unroll
  for (int e = 0; e < 16; ++e) b[e] = (__bf16)p[(k0 + kb_map(e, half)) * ld + n];
  return b;
}
// B(k,n) = W[n*ldw + k] for weight W[N x K] f32 (W^T). UNGUARDED: every GEMM
// width is a multiple of 64 so the tile never runs off the weight matrix.
// Each lane reads two runs of 8 consecutive floats -> vectorizes to clause'd b128 loads.
DEV v16bf frag_b_wt_f32(const float* W, int ldw, int k0, int n0) {
  int lane = threadIdx.x & 31, half = lane >> 4, n = n0 + (lane & 15);
  v16bf b;
#pragma unroll
  for (int e = 0; e < 16; ++e) {
    int k = k0 + kb_map(e, half);
    b[e] = (__bf16)W[(size_t)n * ldw + k];
  }
  return b;
}
// B(k,n) = src[n*ld + k] (bf16 transposed, for G = C @ B^T)
DEV v16bf frag_bt_bf16(const __bf16* p, int ld, int k0, int n0) {
  int lane = threadIdx.x & 31, half = lane >> 4, n = n0 + (lane & 15);
  v16bf b;
#pragma unroll
  for (int e = 0; e < 16; ++e) b[e] = p[n * ld + k0 + kb_map(e, half)];
  return b;
}

DEV float siluf(float x) { return x / (1.f + __expf(-x)); }
DEV float softplusf(float x) { return (x > 20.f) ? x : log1pf(__expf(x)); }

// ================= generic GEMM: C[M x N] = A[M x K] @ W[N x K]^T (+bias)(+res) =====
// grid: (M/64, N/64); block: 128 (4 waves); one 16x64 strip per wave.
// K is a template parameter so the K-loop fully unrolls.
template <int K>
__global__ __launch_bounds__(128) void k_gemm(const float* __restrict__ A,
                                              const float* __restrict__ W,
                                              const float* __restrict__ bias,
                                              const float* __restrict__ res,
                                              float* __restrict__ C, int Ncols) {
  const int lane = threadIdx.x & 31, wid = threadIdx.x >> 5;
  const int half = lane >> 4, l15 = lane & 15;
  const int row0 = (blockIdx.x * 4 + wid) * 16;
  const int n0   = blockIdx.y * 64;

  v8f z = {0.f, 0.f, 0.f, 0.f, 0.f, 0.f, 0.f, 0.f};
  v8f acc[4] = {z, z, z, z};

#pragma unroll
  for (int kk = 0; kk < K; kk += 32) {
    if (kk + 32 < K)  // compile-time under full unroll; emits global_prefetch_b8
      __builtin_prefetch(&A[(size_t)(row0 + l15) * K + kk + 32], 0, 1);
    v16bf a = frag_a_f32(A, K, row0, kk);
#pragma unroll
    for (int t = 0; t < 4; ++t) {
      v16bf b = frag_b_wt_f32(W, K, kk, n0 + t * 16);
      acc[t] = wmma_bf16(a, b, acc[t]);
    }
  }
#pragma unroll
  for (int t = 0; t < 4; ++t) {
#pragma unroll
    for (int r = 0; r < 8; ++r) {
      int m = row0 + r + 8 * half;
      int col = n0 + t * 16 + l15;
      size_t idx = (size_t)m * Ncols + col;
      float v = acc[t][r];
      if (bias) v += bias[col];
      if (res) v += res[idx];
      C[idx] = v;
    }
  }
}

// ================= RMSNorm (optionally input-gated by silu(gate), optional post-res) ==
// one row per wave; grid = rows/4, block 128.
__global__ __launch_bounds__(128) void k_rms(const float* __restrict__ in,
                                             const float* __restrict__ w,
                                             float* __restrict__ out, int rowlen,
                                             const float* __restrict__ gate, int gstride,
                                             const float* __restrict__ res) {
  const int lane = threadIdx.x & 31, wid = threadIdx.x >> 5;
  const int row = blockIdx.x * 4 + wid;
  const float* ip = in + (size_t)row * rowlen;
  const int npl = rowlen >> 5;  // 4 or 8
  float v[8];
  float ss = 0.f;
  for (int j = 0; j < npl; ++j) {
    int c = lane + (j << 5);
    float x = ip[c];
    if (gate) {
      float g = gate[(size_t)row * gstride + c];
      x *= siluf(g);
    }
    v[j] = x;
    ss += x * x;
  }
#pragma unroll
  for (int off = 16; off > 0; off >>= 1) ss += __shfl_xor(ss, off, 32);
  float sc = rsqrtf(ss / (float)rowlen + 1e-5f);
  for (int j = 0; j < npl; ++j) {
    int c = lane + (j << 5);
    float o = v[j] * sc * w[c];
    if (res) o += res[(size_t)row * rowlen + c];
    out[(size_t)row * rowlen + c] = o;
  }
}

// ================= causal depthwise conv (4 taps) + silu on xBC channels ============
// zx has row stride ZXW (768); xBC occupies columns [DINc, DINc+CCHc).
__global__ __launch_bounds__(256) void k_conv(const float* __restrict__ zx,
                                              const float* __restrict__ cw,
                                              const float* __restrict__ cb,
                                              float* __restrict__ xc) {
  int idx = blockIdx.x * 256 + threadIdx.x;  // flat (b,t,c), c fastest
  int c = idx & (CCHc - 1);
  int bt = idx >> 9;  // b*L + t
  int t = bt & (LL - 1);
  float acc = cb[c];
#pragma unroll
  for (int k = 0; k < CONVc; ++k) {
    int tt = t - (CONVc - 1) + k;
    if (tt >= 0)  // wave-uniform (whole wave shares t): cheap scalar branch
      acc += zx[(size_t)(bt - (CONVc - 1) + k) * ZXW + DINc + c] * cw[c * CONVc + k];
  }
  xc[idx] = siluf(acc);
}

// ===== dt head: raw = hn . Wdt[h];  dt = softplus(raw + bias); log(dA) = -dt*e^A ====
// The 2-wide dt projection is a dot product here so the big in_proj GEMM stays an
// exact multiple of 64 (no ragged, guarded WMMA tiles). float4 loads for bandwidth.
__global__ __launch_bounds__(256) void k_dtprep(const float* __restrict__ hn,
                                                const float* __restrict__ Wdt,
                                                const float* __restrict__ dtb,
                                                const float* __restrict__ alog,
                                                float* __restrict__ dtv,
                                                float* __restrict__ slg) {
  int idx = blockIdx.x * 256 + threadIdx.x;  // flat (b,l,h)
  int hh = idx & (HH - 1);
  const float4* hr = (const float4*)(hn + (size_t)(idx >> 1) * DI);
  const float4* wr = (const float4*)(Wdt + hh * DI);
  float raw = 0.f;
#pragma unroll 8
  for (int k = 0; k < DI / 4; ++k) {
    float4 a = hr[k], b = wr[k];
    raw += a.x * b.x + a.y * b.y + a.z * b.z + a.w * b.w;
  }
  float dt = softplusf(raw + dtb[hh]);
  dtv[idx] = dt;
  slg[idx] = -dt * __expf(alog[hh]);
}

// ================= chunked SSD scan (Mamba-2) on WMMA ================================
// One workgroup per (b, h, P-half). State slice Sn[N][64] kept in f32 LDS across 128
// sequential chunks of QC=32 steps; all four per-chunk contractions run on bf16 WMMA.
__global__ __launch_bounds__(256) void k_scan(const float* __restrict__ xc,
                                              const float* __restrict__ dtv,
                                              const float* __restrict__ slg,
                                              const float* __restrict__ Dp,
                                              float* __restrict__ yb) {
  __shared__ __bf16 sX[QC][64];    // x chunk (time x Pslice)
  __shared__ __bf16 sB[QC][NN];    // B chunk (time x N)
  __shared__ __bf16 sC[QC][NN];    // C chunk (time x N)
  __shared__ __bf16 sM[QC][QC];    // masked/decayed intra weights
  __shared__ float  sS[NN][64];    // state^T slice (N x Pslice), f32 across chunks
  __shared__ float  s_s[QC], s_dt[QC], s_eS[QC], s_bs[QC];
  __shared__ float  s_T;

  const int tid = threadIdx.x, lane = tid & 31, wid = tid >> 5;
  const int half = lane >> 4, l15 = lane & 15;
  const int b = blockIdx.x >> 2;
  const int h = (blockIdx.x >> 1) & 1;
  const int ph = blockIdx.x & 1;
  const int p0 = ph * 64;
  const float dval = Dp[h];

  for (int i = tid; i < NN * 64; i += 256) (&sS[0][0])[i] = 0.f;
  __syncthreads();

  for (int cc = 0; cc < LL / QC; ++cc) {
    const int t0 = cc * QC;
    // ---- stage 0: stage chunk into LDS (cvt f32 -> bf16) ----
    for (int idx = tid; idx < QC * 64; idx += 256) {
      int i = idx >> 6, c = idx & 63;
      sX[i][c] = (__bf16)xc[(size_t)(b * LL + t0 + i) * CCHc + h * PP + p0 + c];
    }
    for (int idx = tid; idx < QC * NN; idx += 256) {
      int i = idx >> 7, c = idx & 127;
      size_t base = (size_t)(b * LL + t0 + i) * CCHc;
      sB[i][c] = (__bf16)xc[base + DINc + c];
      sC[i][c] = (__bf16)xc[base + DINc + NN + c];
    }
    if (tid == 0) {  // cumulative log-decays (stable segsum form)
      float run = 0.f;
      for (int i = 0; i < QC; ++i) {
        int g = (b * LL + t0 + i) * HH + h;
        run += slg[g];
        s_s[i] = run;
        s_dt[i] = dtv[g];
      }
      s_T = __expf(run);
      for (int i = 0; i < QC; ++i) {
        s_eS[i] = __expf(s_s[i]);
        s_bs[i] = __expf(run - s_s[i]) * s_dt[i];
      }
    }
    __syncthreads();

    // ---- stage 1: G = C @ B^T (QCxQC, K=N); decay+mask, branch-free ----
    if (wid < 4) {
      int ti = wid >> 1, tj = wid & 1;
      v8f acc = {0.f, 0.f, 0.f, 0.f, 0.f, 0.f, 0.f, 0.f};
#pragma unroll
      for (int kk = 0; kk < NN; kk += 32) {
        v16bf a = frag_a_bf16(&sC[0][0], NN, ti * 16, kk);
        v16bf bm = frag_bt_bf16(&sB[0][0], NN, kk, tj * 16);
        acc = wmma_bf16(a, bm, acc);
      }
      // hoist lane-invariant column terms; mask via multiply (no exec branches)
      const int j = tj * 16 + l15;
      const float sj = s_s[j];
      const float dj = s_dt[j];
#pragma unroll
      for (int r = 0; r < 8; ++r) {
        int i = ti * 16 + r + 8 * half;
        float d = fminf(s_s[i] - sj, 0.f);   // clamp: exp always safe
        float msk = (j <= i) ? 1.f : 0.f;    // v_cndmask, branch-free
        sM[i][j] = (__bf16)(acc[r] * __expf(d) * dj * msk);
      }
    }
    __syncthreads();

    // ---- stage 2: Y = M @ X  +  (e^s * C) @ Sn ; store + D skip ----
    {
      int ti = wid >> 2, tj = wid & 3;
      v8f acc = {0.f, 0.f, 0.f, 0.f, 0.f, 0.f, 0.f, 0.f};
      {  // intra-chunk (K = QC = 32)
        v16bf a = frag_a_bf16(&sM[0][0], QC, ti * 16, 0);
        v16bf bm = frag_b_bf16(&sX[0][0], 64, 0, tj * 16);
        acc = wmma_bf16(a, bm, acc);
      }
#pragma unroll
      for (int kk = 0; kk < NN; kk += 32) {  // inter-chunk (K = N)
        v16bf a = frag_a_bf16_rowscale(&sC[0][0], NN, ti * 16, kk, s_eS);
        v16bf bm = frag_b_f32(&sS[0][0], 64, kk, tj * 16);
        acc = wmma_bf16(a, bm, acc);
      }
#pragma unroll
      for (int r = 0; r < 8; ++r) {
        int i = ti * 16 + r + 8 * half;
        int p = tj * 16 + l15;
        float xv = (float)sX[i][p];
        yb[(size_t)(b * LL + t0 + i) * DINc + h * PP + p0 + p] = acc[r] + dval * xv;
      }
    }
    __syncthreads();

    // ---- stage 3: Sn = T*Sn + (decay*B)^T @ X  (N x Pslice, K = QC) ----
    for (int tt = wid; tt < 32; tt += 8) {
      int ti = tt >> 2, tj = tt & 3;
      v8f acc;
#pragma unroll
      for (int r = 0; r < 8; ++r)
        acc[r] = s_T * sS[ti * 16 + r + 8 * half][tj * 16 + l15];
      v16bf a = frag_at_bf16_kscale(&sB[0][0], NN, ti * 16, 0, s_bs);
      v16bf bm = frag_b_bf16(&sX[0][0], 64, 0, tj * 16);
      acc = wmma_bf16(a, bm, acc);
#pragma unroll
      for (int r = 0; r < 8; ++r)
        sS[ti * 16 + r + 8 * half][tj * 16 + l15] = acc[r];
    }
    __syncthreads();
  }
}

// =====================================================================================
extern "C" void kernel_launch(void* const* d_in, const int* in_sizes, int n_in,
                              void* d_out, int out_size, void* d_ws, size_t ws_size,
                              hipStream_t stream) {
  (void)in_sizes; (void)n_in; (void)out_size; (void)ws_size;
  const float* x       = (const float*)d_in[0];
  const float* in_w    = (const float*)d_in[1];
  const float* in_b    = (const float*)d_in[2];
  const float* out_w   = (const float*)d_in[3];
  const float* out_b   = (const float*)d_in[4];
  const float* l_rms_w = (const float*)d_in[5];
  const float* l_in_w  = (const float*)d_in[6];
  const float* conv_w  = (const float*)d_in[7];
  const float* conv_b  = (const float*)d_in[8];
  const float* dt_bias = (const float*)d_in[9];
  const float* A_log   = (const float*)d_in[10];
  const float* D_p     = (const float*)d_in[11];
  const float* gnorm_w = (const float*)d_in[12];
  const float* l_out_w = (const float*)d_in[13];
  const float* fnorm_w = (const float*)d_in[14];
  float* outp = (float*)d_out;

  // workspace carve (all f32)
  float* w = (float*)d_ws;
  float* xp   = w; w += (size_t)MM * DI;
  float* hbuf = w; w += (size_t)MM * DI;
  float* hn   = w; w += (size_t)MM * DI;
  float* zx   = w; w += (size_t)MM * ZXW;
  float* xcb  = w; w += (size_t)MM * CCHc;
  float* dtvb = w; w += (size_t)MM * HH;
  float* slgb = w; w += (size_t)MM * HH;
  float* ybf  = w; w += (size_t)MM * DINc;
  float* ynb  = w; w += (size_t)MM * DINc;

  const dim3 gb(128);

  // xp = x @ in_w^T + in_b
  k_gemm<NE><<<dim3(MM / 64, DI / 64), gb, 0, stream>>>(x, in_w, in_b, nullptr, xp, DI);
  // h = xp
  hipMemcpyAsync(hbuf, xp, sizeof(float) * (size_t)MM * DI, hipMemcpyDeviceToDevice,
                 stream);

  for (int l = 0; l < NLc; ++l) {
    const float* Wl = l_in_w + (size_t)l * DPROJc * DI;
    // hn = rms(h) * w
    k_rms<<<MM / 4, 128, 0, stream>>>(hbuf, l_rms_w + l * DI, hn, DI, nullptr, 0,
                                      nullptr);
    // zx = hn @ Wl[0:768]^T   (768-wide, exact 64-multiples, no guards)
    k_gemm<DI><<<dim3(MM / 64, ZXW / 64), gb, 0, stream>>>(hn, Wl, nullptr, nullptr, zx,
                                                           ZXW);
    // conv + silu on xBC channels
    k_conv<<<((size_t)MM * CCHc) / 256, 256, 0, stream>>>(zx, conv_w + l * CCHc * CONVc,
                                                          conv_b + l * CCHc, xcb);
    // dt head dot-product + softplus + log-decay
    k_dtprep<<<((size_t)MM * HH) / 256, 256, 0, stream>>>(
        hn, Wl + (size_t)(DINc + CCHc) * DI, dt_bias + l * HH, A_log + l * HH, dtvb,
        slgb);
    // chunked SSD scan (B*H*2 workgroups)
    k_scan<<<BB * HH * 2, 256, 0, stream>>>(xcb, dtvb, slgb, D_p + l * HH, ybf);
    // yn = rms(y * silu(z)) * gnorm_w   (z = zx columns 0..255)
    k_rms<<<MM / 4, 128, 0, stream>>>(ybf, gnorm_w + l * DINc, ynb, DINc, zx, ZXW,
                                      nullptr);
    // h = h + yn @ l_out_w^T
    k_gemm<DINc><<<dim3(MM / 64, DI / 64), gb, 0, stream>>>(
        ynb, l_out_w + (size_t)l * DI * DINc, nullptr, hbuf, hbuf, DI);
  }

  // hn = xp + rms(h) * fnorm_w
  k_rms<<<MM / 4, 128, 0, stream>>>(hbuf, fnorm_w, hn, DI, nullptr, 0, xp);
  // out = hn @ out_w^T + out_b
  k_gemm<DI><<<dim3(MM / 64, DOUTc / 64), gb, 0, stream>>>(hn, out_w, out_b, nullptr,
                                                           outp, DOUTc);
}
